// FFTConv1d_8942121910410
// MI455X (gfx1250) — compile-verified, matrix operand order checked by reading, not run on hardware
//
#include <hip/hip_runtime.h>
#include <hip/hip_bf16.h>
#include <stdint.h>

// ---------------------------------------------------------------------------
// FFT-conv (valid cross-correlation) as implicit GEMM on CDNA5 WMMA.
//   out[b,d,t] = sum_{c,k} sig[b,c,t+k] * w[d,c,k] + bias[d]
// M = Cout = 64, N = 128 t-columns per block (per b), reduction = Cin*K = 65536.
// bf16 fragments, fp32 accumulation via v_wmma_f32_16x16x32_bf16.
// Hankel alignment solved with 8x shift-replicated signal window in LDS.
// Double-buffered LDS stages; weight tiles move via async global->LDS loads
// (ASYNCcnt) when the toolchain exposes the gfx1250 builtins.
// ---------------------------------------------------------------------------

typedef __bf16 v8bf  __attribute__((ext_vector_type(8)));
typedef __bf16 v16bf __attribute__((ext_vector_type(16)));
typedef float  v8f   __attribute__((ext_vector_type(8)));

#define BB      16
#define CIN     64
#define COUT    64
#define LL      32768
#define KW      1024
#define TLEN    31745               // L - K + 1
#define LPAD    34816               // L + 2048 zero pad (edge tiles + shifts)
#define RED     (CIN * KW)          // 65536 reduction length
#define TILE_N  128                 // t columns per block
#define KCH     128                 // reduction chunk staged in LDS
#define NSTAGES (RED / KCH)         // 512
#define NTILES  249                 // ceil(31745 / 128)

#if defined(__AMDGCN__) &&                                                   \
    __has_builtin(__builtin_amdgcn_global_load_async_to_lds_b128) &&         \
    __has_builtin(__builtin_amdgcn_s_wait_asynccnt)
#define USE_ASYNC 1
// Builtin signature (from hipcc diagnostic): param1 = AS(1) v4i*, param2 =
// AS(3) v4i*, then two immediate ints (offset, cpol). v4i is a GCC-style
// vector_size(16) int vector.
typedef int v4i __attribute__((vector_size(16)));
typedef __attribute__((address_space(1))) v4i* gasp;
typedef __attribute__((address_space(3))) v4i* lasp;
#else
#define USE_ASYNC 0
#endif

__device__ __forceinline__ unsigned short f2bf(float f) {
  unsigned int u = __float_as_uint(f);
  u += 0x7FFFu + ((u >> 16) & 1u);      // round-to-nearest-even
  return (unsigned short)(u >> 16);
}

__device__ __forceinline__ v16bf cat8(v8bf lo, v8bf hi) {
  return __builtin_shufflevector(lo, hi, 0, 1, 2, 3, 4, 5, 6, 7, 8, 9, 10, 11,
                                 12, 13, 14, 15);
}

// ---- Phase 1a: signal fp32 -> bf16, rows padded to LPAD with zeros ---------
__global__ void cvt_signal(const float* __restrict__ sig,
                           unsigned short* __restrict__ dst) {
  const size_t total = (size_t)BB * CIN * LPAD;
  for (size_t i = (size_t)blockIdx.x * blockDim.x + threadIdx.x; i < total;
       i += (size_t)gridDim.x * blockDim.x) {
    const size_t row = i / LPAD;
    const int    l   = (int)(i - row * LPAD);
    dst[i] = (l < LL) ? f2bf(sig[row * LL + l]) : (unsigned short)0;
  }
}

// ---- Phase 1b: weight fp32 -> bf16 ([d][c*K+k], reduction-contiguous) ------
__global__ void cvt_weight(const float* __restrict__ w,
                           unsigned short* __restrict__ dst) {
  const size_t total = (size_t)COUT * RED;
  for (size_t i = (size_t)blockIdx.x * blockDim.x + threadIdx.x; i < total;
       i += (size_t)gridDim.x * blockDim.x) {
    dst[i] = f2bf(w[i]);
  }
}

// ---- Phase 2: WMMA implicit GEMM, double-buffered + async staging ----------
__global__ __launch_bounds__(256)
void fftconv_wmma(const unsigned short* __restrict__ sigbf,
                  const unsigned short* __restrict__ wbf,
                  const float* __restrict__ bias,
                  float* __restrict__ out) {
  __shared__ __align__(16) unsigned short wt[2][COUT * KCH]; // 2 x 16 KB
  __shared__ __align__(16) unsigned short sp[2][8 * 256];    // 2 x 4 KB

  const int tid  = threadIdx.x;
  const int lane = tid & 31;
  const int wid  = tid >> 5;
  const int half = lane >> 4;          // fragment K-half select
  const int lr   = lane & 15;
  const int wm   = wid & 3;            // 16-row d tile
  const int wn   = wid >> 2;           // 0/1: 64-wide t group
  const int b    = blockIdx.y;
  const int t0   = blockIdx.x * TILE_N;

  v8f acc[4];
#pragma unroll
  for (int ns = 0; ns < 4; ++ns)
#pragma unroll
    for (int e = 0; e < 8; ++e) acc[ns][e] = 0.0f;

  // staging work decomposition (256 threads)
  const int wd  = tid >> 2;            // weight row 0..63
  const int wk0 = (tid & 3) * 32;      // 32 bf16 = 64 B per thread
  const int sh  = tid >> 5;            // replica shift 0..7
  const int sj  = (tid & 31) * 8;      // 8 bf16 per thread per replica

  const size_t wrow  = (size_t)wd * RED + wk0;
  const size_t sbase = (size_t)b * CIN * LPAD + t0;

  // ---- stage loader: stage st -> buffers[dst] ------------------------------
  auto stage_load = [&](int st, int dst) {
    // weight tile: 4 x b128 per thread, fully coalesced
#if USE_ASYNC
    {
      unsigned short* g =
          const_cast<unsigned short*>(wbf) + wrow + (size_t)st * KCH;
      unsigned short* l = &wt[dst][wd * KCH + wk0];
#pragma unroll
      for (int q = 0; q < 4; ++q)
        __builtin_amdgcn_global_load_async_to_lds_b128(
            (gasp)(g + q * 8), (lasp)(l + q * 8), 0, 0);
    }
#else
    {
      const uint4* g = (const uint4*)(wbf + wrow + (size_t)st * KCH);
      uint4* l = (uint4*)(&wt[dst][wd * KCH + wk0]);
#pragma unroll
      for (int q = 0; q < 4; ++q) l[q] = g[q];
      if (st + 1 < NSTAGES)
        __builtin_prefetch((const void*)(wbf + wrow + (size_t)(st + 1) * KCH),
                           0, 1);
    }
#endif
    // signal window, 8 shift-replicas: sp[dst][sh][i] = sig[row + i + sh]
    {
      const unsigned short* g =
          sigbf + sbase + (size_t)(st >> 3) * LPAD + (st & 7) * KCH + sj + sh;
      unsigned int p0 = (unsigned int)g[0] | ((unsigned int)g[1] << 16);
      unsigned int p1 = (unsigned int)g[2] | ((unsigned int)g[3] << 16);
      unsigned int p2 = (unsigned int)g[4] | ((unsigned int)g[5] << 16);
      unsigned int p3 = (unsigned int)g[6] | ((unsigned int)g[7] << 16);
      *(uint4*)(&sp[dst][(sh << 8) + sj]) = make_uint4(p0, p1, p2, p3);
    }
  };

  // ---- software pipeline: load(st+1) overlaps compute(st) ------------------
  stage_load(0, 0);
#if USE_ASYNC
  __builtin_amdgcn_s_wait_asynccnt(0);
#endif
  __syncthreads();

  for (int st = 0; st < NSTAGES; ++st) {
    const int buf = st & 1;
    if (st + 1 < NSTAGES) stage_load(st + 1, buf ^ 1);

    const unsigned short* wA = &wt[buf][0];
    const unsigned short* sB = &sp[buf][0];
#pragma unroll
    for (int ks = 0; ks < 4; ++ks) {             // 4 x K=32 steps per chunk
      const int k0 = ks * 32;
      // A fragment (weight 16x32): lane holds row M=lr, K = 8*half+[0,8) and 16+8*half+[0,8)
      const v8bf* ap = (const v8bf*)(&wA[(wm * 16 + lr) * KCH + k0 + half * 8]);
      const v16bf afrag = cat8(ap[0], ap[2]);
#pragma unroll
      for (int ns = 0; ns < 4; ++ns) {           // 4 x N=16 subtiles per wave
        // B element (kk, n) = window[n + kk]; 8 contiguous bf16 at x ->
        // 16B-aligned ds_load_b128 from shift-replica (x & 7).
        const int x = wn * 64 + ns * 16 + lr + k0 + half * 8;
        const int s = x & 7;
        const v8bf* bp = (const v8bf*)(&sB[(s << 8) + (x - s)]);
        const v16bf bfrag = cat8(bp[0], bp[2]);
        acc[ns] = __builtin_amdgcn_wmma_f32_16x16x32_bf16(
            false, afrag, false, bfrag, (short)0, acc[ns], false, false);
      }
    }

#if USE_ASYNC
    __builtin_amdgcn_s_wait_asynccnt(0);  // next-stage tiles resident in LDS
#endif
    __syncthreads();                      // publish buf^1, retire reads of buf
  }

  // -- epilogue: C/D layout VGPR v -> M = v + 8*(lane/16), N = lane%16 -------
#pragma unroll
  for (int ns = 0; ns < 4; ++ns) {
    const int t = t0 + wn * 64 + ns * 16 + lr;
    if (t < TLEN) {
#pragma unroll
      for (int v = 0; v < 8; ++v) {
        const int d = wm * 16 + half * 8 + v;
        out[((size_t)b * COUT + d) * TLEN + t] = acc[ns][v] + bias[d];
      }
    }
  }
}

// ---------------------------------------------------------------------------
extern "C" void kernel_launch(void* const* d_in, const int* in_sizes, int n_in,
                              void* d_out, int out_size, void* d_ws,
                              size_t ws_size, hipStream_t stream) {
  const float* sig  = (const float*)d_in[0];  // [16,64,32768]
  const float* w    = (const float*)d_in[1];  // [64,64,1024]
  const float* bias = (const float*)d_in[2];  // [64,1]
  float* out = (float*)d_out;                 // [16,64,31745]

  unsigned short* sig_bf = (unsigned short*)d_ws;               // 71.3 MB
  unsigned short* w_bf   = sig_bf + (size_t)BB * CIN * LPAD;    // +8.4 MB

  cvt_signal<<<4096, 256, 0, stream>>>(sig, sig_bf);
  cvt_weight<<<2048, 256, 0, stream>>>(w, w_bf);

  dim3 grid(NTILES, BB);
  fftconv_wmma<<<grid, 256, 0, stream>>>(sig_bf, w_bf, bias, out);
}
